// RSSM_Teacher_46480136077862
// MI455X (gfx1250) — matrix operand order repaired and checked by reading, not compiled, not run on hardware
//
#include <hip/hip_runtime.h>
#include <hip/hip_bf16.h>
#include <math.h>

// ---------------- problem dims ----------------
#define B_      512
#define T_      64
#define STOCH   30
#define DETER   200
#define HIDDEN  200
#define NA      12
#define EMBED   1024
#define BT      (B_ * T_)
#define OUTW    (6 * STOCH + DETER)   // 380

// ---------------- WMMA types ----------------
typedef __attribute__((ext_vector_type(16))) __bf16 v16bf;
typedef __attribute__((ext_vector_type(8)))  float  v8f;

// packed-weight offsets in workspace (ushort/bf16 units)
#define SZ_WI     (32   * 208)
#define SZ_WG     (224  * 608)
#define SZ_W200   (224  * 208)
#define SZ_W60    (224  * 64)
#define SZ_WOOE   (1024 * 208)
#define OFF_WI    0
#define OFF_WGX   (OFF_WI   + SZ_WI)
#define OFF_WGD   (OFF_WGX  + SZ_WG)
#define OFF_WIO   (OFF_WGD  + SZ_WG)
#define OFF_WIMS  (OFF_WIO  + SZ_W200)
#define OFF_WOOD  (OFF_WIMS + SZ_W60)
#define OFF_WOBS  (OFF_WOOD + SZ_W200)
#define OFF_WOOE  (OFF_WOBS + SZ_W60)

// byte offsets for f32 precompute buffers in workspace
#define WSOFF_XPRE    (2u * 1024u * 1024u)                         // after ~1.2MB packed weights
#define WSOFF_OBSPRE  (WSOFF_XPRE + (size_t)BT * 200u * 4u)

// ---------------- helpers ----------------
__device__ __forceinline__ unsigned short f2bfu(float f) {
    union { float f; unsigned u; } x; x.f = f;
    unsigned r = (x.u + 0x7FFFu + ((x.u >> 16) & 1u)) >> 16;
    return (unsigned short)r;
}
__device__ __forceinline__ float bf2f(unsigned short u) {
    union { unsigned u; float f; } y; y.u = ((unsigned)u) << 16;
    return y.f;
}
__device__ __forceinline__ float sigmoidf_(float x) { return 1.0f / (1.0f + __expf(-x)); }
__device__ __forceinline__ float eluf_(float x)     { return x > 0.0f ? x : expm1f(x); }
__device__ __forceinline__ float softplusf_(float x){ return x > 20.0f ? x : log1pf(expf(x)); }

struct U16x2 { uint4 lo, hi; };   // 32 bytes -> v16bf

// A fragment: 16x32 in contiguous k order (B uses the same). lanes 0-15: row M=lane, K {k0..k0+15};
// lanes 16-31: row M=lane-16, K {k0+16..k0+31}. Two 16B LDS loads.
__device__ __forceinline__ v16bf load_a_frag(const unsigned short* act, int astride, int k0, int lane) {
    int lanelo = lane & 15;
    int hi     = lane >> 4;
    const unsigned short* ar = act + lanelo * astride + k0 + hi * 16;
    U16x2 t;
    t.lo = *(const uint4*)(ar);
    t.hi = *(const uint4*)(ar + 8);
    return __builtin_bit_cast(v16bf, t);
}
// B fragment: packed transposed [Npad][Kpad] bf16; lane = column, K pairs per VGPR,
// lane-hi selects K+16. One 32B load.
__device__ __forceinline__ v16bf load_b_frag(const unsigned short* wp, int kpad, int n0, int k0, int lane) {
    int lanelo = lane & 15;
    int hi     = lane >> 4;
    return *(const v16bf*)(wp + (size_t)(n0 + lanelo) * kpad + k0 + hi * 16);
}

#define WMMA_BF16(a, b, d) \
    __builtin_amdgcn_wmma_f32_16x16x32_bf16(false, (a), false, (b), (short)0, (d), false, false)

// single-tile accumulate, software-pipelined B loads
__device__ __forceinline__ v8f gemm_acc(v8f d, const unsigned short* act, int astride,
                                        const unsigned short* wp, int kpad, int nchunks,
                                        int n0, int lane) {
    v16bf b = load_b_frag(wp, kpad, n0, 0, lane);
    for (int c = 0; c < nchunks; ++c) {
        v16bf a  = load_a_frag(act, astride, c * 32, lane);
        int  cn  = (c + 1 < nchunks) ? c + 1 : c;
        v16bf nb = load_b_frag(wp, kpad, n0, cn * 32, lane);
        d = WMMA_BF16(a, b, d);
        b = nb;
    }
    return d;
}
// dual-tile accumulate: one A fragment feeds two independent WMMA chains,
// B loads for chunk c+1 prefetched before the chunk-c WMMAs.
__device__ __forceinline__ void gemm_acc2(v8f& d0, v8f& d1,
                                          const unsigned short* act, int astride,
                                          const unsigned short* wp, int kpad, int nchunks,
                                          int n0, int n1, int lane) {
    v16bf b0 = load_b_frag(wp, kpad, n0, 0, lane);
    v16bf b1 = load_b_frag(wp, kpad, n1, 0, lane);
    for (int c = 0; c < nchunks; ++c) {
        v16bf a   = load_a_frag(act, astride, c * 32, lane);
        int  cn   = (c + 1 < nchunks) ? c + 1 : c;
        v16bf nb0 = load_b_frag(wp, kpad, n0, cn * 32, lane);
        v16bf nb1 = load_b_frag(wp, kpad, n1, cn * 32, lane);
        d0 = WMMA_BF16(a, b0, d0);
        d1 = WMMA_BF16(a, b1, d1);
        b0 = nb0; b1 = nb1;
    }
}

// ---------------- kernel 1: pack a weight matrix transposed+padded to bf16 ----------------
__global__ void pack_kernel(const float* __restrict__ src, int row0, int ld, int K, int N,
                            unsigned short* __restrict__ dst, int kpad, int total) {
    int idx = blockIdx.x * blockDim.x + threadIdx.x;
    if (idx >= total) return;
    int n = idx / kpad, k = idx % kpad;
    float v = (k < K && n < N) ? src[(size_t)(row0 + k) * ld + n] : 0.0f;
    dst[idx] = f2bfu(v);
}

// ---------------- kernel 2a: x_pre = action @ inp_W[30:42] + inp_b ----------------
__global__ void xpre_kernel(const float* __restrict__ action, const float* __restrict__ inpW,
                            const float* __restrict__ inpB, float* __restrict__ xpre) {
    int idx = blockIdx.x * blockDim.x + threadIdx.x;
    if (idx >= BT * 200) return;
    int n = idx % 200, bt = idx / 200;
    float s = inpB[n];
#pragma unroll
    for (int a = 0; a < NA; ++a)
        s += action[(size_t)bt * NA + a] * inpW[(size_t)(STOCH + a) * 200 + n];
    xpre[idx] = s;
}

// ---------------- kernel 2b: obs_pre = embed @ obs_out_W[200:1224] + obs_out_b (WMMA) ----------------
__global__ __launch_bounds__(256)
void obspre_kernel(const float* __restrict__ embed, const unsigned short* __restrict__ wp,
                   const float* __restrict__ obsOutB, float* __restrict__ obspre) {
    __shared__ __align__(16) unsigned short emb[16 * 1024];
    int bt0 = blockIdx.x * 16;
    for (int idx = threadIdx.x; idx < 16 * 1024; idx += 256)
        emb[idx] = f2bfu(embed[(size_t)(bt0 + (idx >> 10)) * EMBED + (idx & 1023)]);
    __syncthreads();

    int wave = threadIdx.x >> 5, lane = threadIdx.x & 31;
    int lanelo = lane & 15, hi8 = (lane >> 4) * 8;
    const unsigned short* wooe = wp + OFF_WOOE;

    auto epi = [&](v8f d, int n0) {
        int col = n0 + lanelo;
        if (col < 200) {
            float bias = obsOutB[col];
#pragma unroll
            for (int i = 0; i < 8; ++i)
                obspre[(size_t)(bt0 + hi8 + i) * 200 + col] = d[i] + bias;
        }
    };
    for (int base = wave * 2; base < 13; base += 16) {
        v8f d0 = {};
        if (base + 1 < 13) {
            v8f d1 = {};
            gemm_acc2(d0, d1, emb, 1024, wooe, 1024, 32, base * 16, (base + 1) * 16, lane);
            epi(d0, base * 16);
            epi(d1, (base + 1) * 16);
        } else {
            d0 = gemm_acc(d0, emb, 1024, wooe, 1024, 32, base * 16, lane);
            epi(d0, base * 16);
        }
    }
}

// ---------------- kernel 3: persistent recurrent RSSM (batch tile = 16 rows) ----------------
__global__ __launch_bounds__(256)
void rssm_kernel(const float* __restrict__ xpre, const float* __restrict__ obspre,
                 const unsigned short* __restrict__ wp,
                 const float* __restrict__ gruB, const float* __restrict__ imgB,
                 const float* __restrict__ imsB, const float* __restrict__ obsB,
                 const float* __restrict__ nzPr, const float* __restrict__ nzPo,
                 float* __restrict__ out) {
    // LDS pool (45.25 KB): overlays reuse the dead `parts` region for h/stat/stoch.
    __shared__ __align__(16) unsigned char smem[46336];
    unsigned short* x_bf  = (unsigned short*)(smem + 0);       // [16][224] bf16 x
    unsigned short* d_bf  = (unsigned short*)(smem + 7168);    // [16][224] bf16 deter
    float*          deter = (float*)(smem + 14336);            // [16][200] f32 state
    unsigned short* parts = (unsigned short*)(smem + 27136);   // [16][600] bf16 GRU parts
    unsigned short* h_bf  = (unsigned short*)(smem + 27136);   // [16][224] overlay
    float*          statf = (float*)(smem + 34304);            // [16][64]  overlay
    unsigned short* st_bf = (unsigned short*)(smem + 38400);   // [16][32]  overlay (stoch)

    for (int i = threadIdx.x; i < 46336 / 4; i += 256) ((unsigned*)smem)[i] = 0u;
    __syncthreads();

    const int b0 = blockIdx.x * 16;
    const int wave = threadIdx.x >> 5, lane = threadIdx.x & 31;
    const int lanelo = lane & 15, hi8 = (lane >> 4) * 8;

    for (int t = 0; t < T_; ++t) {
        // ---- S1: x = elu(stoch @ Wi_s + x_pre) ----
        {
            auto epi = [&](v8f d, int n0) {
                int col = n0 + lanelo;
                if (col < 200) {
#pragma unroll
                    for (int i = 0; i < 8; ++i) {
                        int row = hi8 + i;
                        float v = d[i] + xpre[((size_t)(b0 + row) * T_ + t) * 200 + col];
                        x_bf[row * 224 + col] = f2bfu(eluf_(v));
                    }
                }
            };
            for (int base = wave * 2; base < 13; base += 16) {
                v8f d0 = {};
                if (base + 1 < 13) {
                    v8f d1 = {};
                    gemm_acc2(d0, d1, st_bf, 32, wp + OFF_WI, 32, 1, base * 16, (base + 1) * 16, lane);
                    epi(d0, base * 16); epi(d1, (base + 1) * 16);
                } else {
                    d0 = gemm_acc(d0, st_bf, 32, wp + OFF_WI, 32, 1, base * 16, lane);
                    epi(d0, base * 16);
                }
            }
        }
        __syncthreads();
        // ---- S2: parts = x @ Wg[0:200] + deter @ Wg[200:400] + gru_b ----
        {
            auto epi = [&](v8f d, int n0) {
                int col = n0 + lanelo;
                if (col < 600) {
                    float bias = gruB[col];
#pragma unroll
                    for (int i = 0; i < 8; ++i) parts[(hi8 + i) * 600 + col] = f2bfu(d[i] + bias);
                }
            };
            for (int base = wave * 2; base < 38; base += 16) {
                v8f d0 = {}, d1 = {};
                gemm_acc2(d0, d1, x_bf, 224, wp + OFF_WGX, 224, 7, base * 16, (base + 1) * 16, lane);
                gemm_acc2(d0, d1, d_bf, 224, wp + OFF_WGD, 224, 7, base * 16, (base + 1) * 16, lane);
                epi(d0, base * 16); epi(d1, (base + 1) * 16);
            }
        }
        __syncthreads();
        // ---- S3: GRU elementwise update (f32 state) ----
        for (int idx = threadIdx.x; idx < 16 * 200; idx += 256) {
            int r = idx / 200, c = idx % 200;
            float reset = sigmoidf_(bf2f(parts[r * 600 + c]));
            float cand  = tanhf(reset * bf2f(parts[r * 600 + 200 + c]));
            float upd   = sigmoidf_(bf2f(parts[r * 600 + 400 + c]) - 1.0f);
            float dn = upd * cand + (1.0f - upd) * deter[r * 200 + c];
            deter[r * 200 + c] = dn;
            d_bf[r * 224 + c]  = f2bfu(dn);
            out[((size_t)(b0 + r) * T_ + t) * OUTW + 180 + c] = dn;
        }
        __syncthreads();
        // ---- S4: h = elu(deter_new @ img_out_W + b) ----
        {
            auto epi = [&](v8f d, int n0) {
                int col = n0 + lanelo;
                if (col < 200) {
                    float bias = imgB[col];
#pragma unroll
                    for (int i = 0; i < 8; ++i) h_bf[(hi8 + i) * 224 + col] = f2bfu(eluf_(d[i] + bias));
                }
            };
            for (int base = wave * 2; base < 13; base += 16) {
                v8f d0 = {};
                if (base + 1 < 13) {
                    v8f d1 = {};
                    gemm_acc2(d0, d1, d_bf, 224, wp + OFF_WIO, 224, 7, base * 16, (base + 1) * 16, lane);
                    epi(d0, base * 16); epi(d1, (base + 1) * 16);
                } else {
                    d0 = gemm_acc(d0, d_bf, 224, wp + OFF_WIO, 224, 7, base * 16, lane);
                    epi(d0, base * 16);
                }
            }
        }
        __syncthreads();
        // ---- S5: prior stats = h @ ims_W + ims_b ----
        {
            auto epi = [&](v8f d, int n0) {
                int col = n0 + lanelo;
                if (col < 60) {
                    float bias = imsB[col];
#pragma unroll
                    for (int i = 0; i < 8; ++i) statf[(hi8 + i) * 64 + col] = d[i] + bias;
                }
            };
            for (int base = wave * 2; base < 4; base += 16) {
                v8f d0 = {}, d1 = {};
                gemm_acc2(d0, d1, h_bf, 224, wp + OFF_WIMS, 224, 7, base * 16, (base + 1) * 16, lane);
                epi(d0, base * 16); epi(d1, (base + 1) * 16);
            }
        }
        __syncthreads();
        for (int idx = threadIdx.x; idx < 16 * STOCH; idx += 256) {
            int r = idx / STOCH, c = idx % STOCH;
            float mean = statf[r * 64 + c];
            float std  = softplusf_(statf[r * 64 + STOCH + c]) + 0.1f;
            float nz   = nzPr[((size_t)(b0 + r) * T_ + t) * STOCH + c];
            float* base = out + ((size_t)(b0 + r) * T_ + t) * OUTW;
            base[90 + c]  = mean + std * nz;   // stoch_pr
            base[120 + c] = mean;              // mean_pr
            base[150 + c] = std;               // std_pr
        }
        __syncthreads();
        // ---- S6: ho = elu(deter_new @ obs_out_W[0:200] + obs_pre) ----
        {
            auto epi = [&](v8f d, int n0) {
                int col = n0 + lanelo;
                if (col < 200) {
#pragma unroll
                    for (int i = 0; i < 8; ++i) {
                        int row = hi8 + i;
                        float v = d[i] + obspre[((size_t)(b0 + row) * T_ + t) * 200 + col];
                        h_bf[row * 224 + col] = f2bfu(eluf_(v));
                    }
                }
            };
            for (int base = wave * 2; base < 13; base += 16) {
                v8f d0 = {};
                if (base + 1 < 13) {
                    v8f d1 = {};
                    gemm_acc2(d0, d1, d_bf, 224, wp + OFF_WOOD, 224, 7, base * 16, (base + 1) * 16, lane);
                    epi(d0, base * 16); epi(d1, (base + 1) * 16);
                } else {
                    d0 = gemm_acc(d0, d_bf, 224, wp + OFF_WOOD, 224, 7, base * 16, lane);
                    epi(d0, base * 16);
                }
            }
        }
        __syncthreads();
        // ---- S7: posterior stats = ho @ obs_W + obs_b ----
        {
            auto epi = [&](v8f d, int n0) {
                int col = n0 + lanelo;
                if (col < 60) {
                    float bias = obsB[col];
#pragma unroll
                    for (int i = 0; i < 8; ++i) statf[(hi8 + i) * 64 + col] = d[i] + bias;
                }
            };
            for (int base = wave * 2; base < 4; base += 16) {
                v8f d0 = {}, d1 = {};
                gemm_acc2(d0, d1, h_bf, 224, wp + OFF_WOBS, 224, 7, base * 16, (base + 1) * 16, lane);
                epi(d0, base * 16); epi(d1, (base + 1) * 16);
            }
        }
        __syncthreads();
        for (int idx = threadIdx.x; idx < 16 * STOCH; idx += 256) {
            int r = idx / STOCH, c = idx % STOCH;
            float mean = statf[r * 64 + c];
            float std  = softplusf_(statf[r * 64 + STOCH + c]) + 0.1f;
            float nz   = nzPo[((size_t)(b0 + r) * T_ + t) * STOCH + c];
            float po   = mean + std * nz;
            float* base = out + ((size_t)(b0 + r) * T_ + t) * OUTW;
            base[c]      = po;                 // stoch_po
            base[30 + c] = mean;               // mean_po
            base[60 + c] = std;                // std_po
            st_bf[r * 32 + c] = f2bfu(po);     // recurrent stoch for next step
        }
        __syncthreads();
    }
}

// ---------------- launch ----------------
extern "C" void kernel_launch(void* const* d_in, const int* in_sizes, int n_in,
                              void* d_out, int out_size, void* d_ws, size_t ws_size,
                              hipStream_t stream) {
    (void)in_sizes; (void)n_in; (void)out_size; (void)ws_size;
    const float* embed   = (const float*)d_in[0];
    const float* action  = (const float*)d_in[1];
    const float* nzPr    = (const float*)d_in[2];
    const float* nzPo    = (const float*)d_in[3];
    const float* inpW    = (const float*)d_in[4];
    const float* inpB    = (const float*)d_in[5];
    const float* gruW    = (const float*)d_in[6];
    const float* gruB    = (const float*)d_in[7];
    const float* imgW    = (const float*)d_in[8];
    const float* imgB    = (const float*)d_in[9];
    const float* imsW    = (const float*)d_in[10];
    const float* imsB    = (const float*)d_in[11];
    const float* obsOutW = (const float*)d_in[12];
    const float* obsOutB = (const float*)d_in[13];
    const float* obsW    = (const float*)d_in[14];
    const float* obsB    = (const float*)d_in[15];

    unsigned short* wp = (unsigned short*)d_ws;
    float* xpre   = (float*)((char*)d_ws + WSOFF_XPRE);
    float* obspre = (float*)((char*)d_ws + WSOFF_OBSPRE);

    auto pack = [&](const float* src, int row0, int ld, int K, int N, int off, int kpad, int npad) {
        int total = kpad * npad;
        pack_kernel<<<(total + 255) / 256, 256, 0, stream>>>(src, row0, ld, K, N, wp + off, kpad, total);
    };
    pack(inpW,    0,   200,  30,  200, OFF_WI,   32,   208);
    pack(gruW,    0,   600,  200, 600, OFF_WGX,  224,  608);
    pack(gruW,    200, 600,  200, 600, OFF_WGD,  224,  608);
    pack(imgW,    0,   200,  200, 200, OFF_WIO,  224,  208);
    pack(imsW,    0,   60,   200, 60,  OFF_WIMS, 224,  64);
    pack(obsOutW, 0,   200,  200, 200, OFF_WOOD, 224,  208);
    pack(obsW,    0,   60,   200, 60,  OFF_WOBS, 224,  64);
    pack(obsOutW, 200, 200, 1024, 200, OFF_WOOE, 1024, 208);

    xpre_kernel<<<(BT * 200 + 255) / 256, 256, 0, stream>>>(action, inpW, inpB, xpre);
    obspre_kernel<<<BT / 16, 256, 0, stream>>>(embed, wp, obsOutB, obspre);
    rssm_kernel<<<B_ / 16, 256, 0, stream>>>(xpre, obspre, wp, gruB, imgB, imsB, obsB,
                                             nzPr, nzPo, (float*)d_out);
}
